// GroupedQueryAttention_69088843924220
// MI455X (gfx1250) — matrix operand order repaired and checked by reading, not compiled, not run on hardware
//
#include <hip/hip_runtime.h>
#include <hip/hip_bf16.h>

// ---------------- problem constants ----------------
static constexpr int S    = 2048;
static constexpr int E    = 2048;
static constexpr int H    = 16;
static constexpr int G    = 4;
static constexpr int HD   = 128;
static constexpr int QDIM = H * HD;   // 2048
static constexpr int KVD  = G * HD;   // 512
static constexpr int OUTD = QDIM;     // 2048

typedef __attribute__((ext_vector_type(16))) __bf16 bf16x16;
typedef __attribute__((ext_vector_type(8)))  float  f32x8;

// f32 -> bf16 round-to-nearest-even
static __device__ __forceinline__ unsigned short f2bf(float f) {
  unsigned u = __builtin_bit_cast(unsigned, f);
  unsigned r = (u + 0x7FFFu + ((u >> 16) & 1u)) >> 16;
  return (unsigned short)r;
}

// ---------------- f32 -> bf16 convert (vec4) ----------------
__global__ __launch_bounds__(256) void cvt_f32_to_bf16(const float* __restrict__ s,
                                                       unsigned short* __restrict__ d,
                                                       int n4) {
  int i = blockIdx.x * 256 + threadIdx.x;
  if (i < n4) {
    float4 v = ((const float4*)s)[i];
    uint2 o;
    o.x = (unsigned)f2bf(v.x) | ((unsigned)f2bf(v.y) << 16);
    o.y = (unsigned)f2bf(v.z) | ((unsigned)f2bf(v.w) << 16);
    ((uint2*)d)[i] = o;
  }
}

// ---------------- RoPE + bf16 convert ----------------
// src: [S, heads*HD] f32, cos/sin: [S, HD], dst: [S, heads*HD] bf16
__global__ __launch_bounds__(256) void rope_bf16(const float* __restrict__ src,
                                                 const float* __restrict__ cosT,
                                                 const float* __restrict__ sinT,
                                                 unsigned short* __restrict__ dst,
                                                 int heads) {
  int idx = blockIdx.x * 256 + threadIdx.x;            // element over S*heads*HD
  int d   = idx % HD;
  int t   = idx / HD;
  int h   = t % heads;
  int s   = t / heads;
  size_t row = (size_t)s * (heads * HD) + (size_t)h * HD;
  float x  = src[row + d];
  float c  = cosT[(size_t)s * HD + d];
  float sn = sinT[(size_t)s * HD + d];
  float rot = (d < HD / 2) ? -src[row + d + HD / 2] : src[row + d - HD / 2];
  dst[row + d] = f2bf(x * c + rot * sn);
}

// ---------------- fragment loads from LDS ([row][K], K-innermost, pitch 40) ----------------
// A 16x32 bf16 layout: lanes<16 kb=0, lanes>=16 kb=8; V0..3 -> K=kb+2v, V4..7 -> K=kb+8+2v
// B 32x16 bf16 layout (stored [col][K]): lanes<16 kb=0, lanes>=16 kb=8; V0..3 -> K=kb+2v, V4..7 -> K=kb+8+2v
template<bool APAT>
static __device__ __forceinline__ bf16x16 load_frag(const unsigned short (*T)[40],
                                                    int row, int lane) {
  bf16x16 f;
  int kb = (lane < 16) ? 0 : 8;
#pragma unroll
  for (int v = 0; v < 8; ++v) {
    int k2 = (v < 4) ? (kb + 2 * v) : (kb + 8 + 2 * v);
    unsigned p = *(const unsigned*)&T[row][k2];
    f[2 * v]     = __builtin_bit_cast(__bf16, (unsigned short)(p & 0xFFFFu));
    f[2 * v + 1] = __builtin_bit_cast(__bf16, (unsigned short)(p >> 16));
  }
  return f;
}

// ---------------- tiled WMMA GEMM: C[z] = A[z] @ B[z>>shiftB]  (f32 accum) ----------------
// Block tile 128x128, K-tile 32, double-buffered LDS with register staging.
// 8 waves; wave (wm,wn) owns rows wm*32..+31, cols wn*64..+63 (8 accumulators).
// A_F32: A source is f32, converted to bf16 while staging to LDS.
// B_TRANS: logical B[k][n] = Bsrc[n*ldb + k] (i.e. Bsrc holds B^T row-major).
template<bool A_F32, bool B_TRANS>
__global__ __launch_bounds__(256) void gemm_wmma(
    const void* __restrict__ Ap, const unsigned short* __restrict__ Bp,
    float* __restrict__ Cp, int K, int lda, int ldb, int ldc,
    long long strideA, long long strideB, long long strideC, int shiftB) {

  __shared__ unsigned short As[2][128][40];   // [buf][m][k]  (2 x 10 KB)
  __shared__ unsigned short Bs[2][128][40];   // [buf][n][k]  (2 x 10 KB)

  const int tid  = threadIdx.x;
  const int lane = tid & 31;
  const int wave = tid >> 5;
  const int wm   = wave & 3;   // 0..3 -> 32-row slab
  const int wn   = wave >> 2;  // 0..1 -> 64-col slab
  const int lrow = lane & 15;

  const int m0 = blockIdx.y * 128;
  const int n0 = blockIdx.x * 128;
  const int z  = blockIdx.z;

  const float*          Af = A_F32 ? ((const float*)Ap) + (size_t)z * strideA : nullptr;
  const unsigned short* Ab = A_F32 ? nullptr : ((const unsigned short*)Ap) + (size_t)z * strideA;
  const unsigned short* B  = Bp + (size_t)(z >> shiftB) * strideB;
  float*                C  = Cp + (size_t)z * strideC;

  // register staging for the next k-tile (4 chunks of 4 elements per thread, per matrix)
  float4 aF[4];
  uint2  aB[4];
  uint2  bR[4];

  auto loadTiles = [&](int k0) {
#pragma unroll
    for (int it = 0; it < 4; ++it) {
      int g   = tid + it * 256;          // 1024 groups of 4 elements
      int row = g >> 3;
      int cg  = (g & 7) * 4;
      if constexpr (A_F32)
        aF[it] = *(const float4*)(Af + (size_t)(m0 + row) * lda + k0 + cg);
      else
        aB[it] = *(const uint2*)(Ab + (size_t)(m0 + row) * lda + k0 + cg);
    }
#pragma unroll
    for (int it = 0; it < 4; ++it) {
      int g = tid + it * 256;
      if constexpr (B_TRANS) {
        int n  = g >> 3;
        int cg = (g & 7) * 4;
        bR[it] = *(const uint2*)(B + (size_t)(n0 + n) * ldb + k0 + cg);
      } else {
        int kk = g >> 5;
        int ng = (g & 31) * 4;
        bR[it] = *(const uint2*)(B + (size_t)(k0 + kk) * ldb + n0 + ng);
      }
    }
  };

  auto storeTiles = [&](int buf) {
#pragma unroll
    for (int it = 0; it < 4; ++it) {
      int g   = tid + it * 256;
      int row = g >> 3;
      int cg  = (g & 7) * 4;
      if constexpr (A_F32) {
        unsigned short* d = &As[buf][row][cg];
        d[0] = f2bf(aF[it].x); d[1] = f2bf(aF[it].y);
        d[2] = f2bf(aF[it].z); d[3] = f2bf(aF[it].w);
      } else {
        *(uint2*)&As[buf][row][cg] = aB[it];
      }
    }
#pragma unroll
    for (int it = 0; it < 4; ++it) {
      int g = tid + it * 256;
      if constexpr (B_TRANS) {
        int n  = g >> 3;
        int cg = (g & 7) * 4;
        *(uint2*)&Bs[buf][n][cg] = bR[it];
      } else {
        int kk = g >> 5;
        int ng = (g & 31) * 4;
        Bs[buf][ng + 0][kk] = (unsigned short)(bR[it].x & 0xFFFFu);
        Bs[buf][ng + 1][kk] = (unsigned short)(bR[it].x >> 16);
        Bs[buf][ng + 2][kk] = (unsigned short)(bR[it].y & 0xFFFFu);
        Bs[buf][ng + 3][kk] = (unsigned short)(bR[it].y >> 16);
      }
    }
  };

  f32x8 acc[2][4] = {};

  const int nk = K >> 5;                 // K / 32, K always a multiple of 32 here
  loadTiles(0);
  storeTiles(0);
  __syncthreads();

  for (int kt = 0; kt < nk; ++kt) {
    const int cur = kt & 1;
    const bool more = (kt + 1) < nk;
    if (more) loadTiles((kt + 1) << 5);   // global loads in flight during WMMA

    bf16x16 afrag[2];
    afrag[0] = load_frag<true>(As[cur], wm * 32 + lrow, lane);
    afrag[1] = load_frag<true>(As[cur], wm * 32 + 16 + lrow, lane);
#pragma unroll
    for (int ni = 0; ni < 4; ++ni) {
      bf16x16 bfrag = load_frag<false>(Bs[cur], wn * 64 + ni * 16 + lrow, lane);
#pragma unroll
      for (int mi = 0; mi < 2; ++mi) {
        acc[mi][ni] = __builtin_amdgcn_wmma_f32_16x16x32_bf16(
            false, afrag[mi], false, bfrag, (short)0, acc[mi][ni], false, false);
      }
    }

    if (more) storeTiles(cur ^ 1);        // fill the other buffer
    __syncthreads();                      // one barrier per k-tile
  }

  // ---- write C per 16x16 f32 D layout ----
#pragma unroll
  for (int mi = 0; mi < 2; ++mi)
#pragma unroll
    for (int ni = 0; ni < 4; ++ni)
#pragma unroll
      for (int v = 0; v < 8; ++v) {
        int r = m0 + wm * 32 + mi * 16 + ((lane < 16) ? v : v + 8);
        int c = n0 + wn * 64 + ni * 16 + (lane & 15);
        C[(size_t)r * ldc + c] = acc[mi][ni][v];
      }
}

// ---------------- masked scaled softmax, in place over p[h][q][:] ----------------
__global__ __launch_bounds__(256) void softmax_mask_kernel(float* __restrict__ p) {
  const int q = blockIdx.x, h = blockIdx.y;
  float* row = p + ((size_t)h * S + q) * S;
  const float scale = 0.08838834764831845f;  // 1/sqrt(128)
  __shared__ float red[8];
  const int tid = threadIdx.x, lane = tid & 31, w = tid >> 5;

  float m = -3.0e38f;
  for (int k = tid; k <= q; k += 256) m = fmaxf(m, row[k] * scale);
#pragma unroll
  for (int o = 16; o; o >>= 1) m = fmaxf(m, __shfl_xor(m, o, 32));
  if (lane == 0) red[w] = m;
  __syncthreads();
#pragma unroll
  for (int i = 0; i < 8; ++i) m = fmaxf(m, red[i]);
  __syncthreads();

  float s = 0.0f;
  for (int k = tid; k <= q; k += 256) s += __expf(row[k] * scale - m);
#pragma unroll
  for (int o = 16; o; o >>= 1) s += __shfl_xor(s, o, 32);
  if (lane == 0) red[w] = s;
  __syncthreads();
  s = 0.0f;
#pragma unroll
  for (int i = 0; i < 8; ++i) s += red[i];
  float inv = 1.0f / s;

  for (int k = tid; k < S; k += 256)
    row[k] = (k <= q) ? __expf(row[k] * scale - m) * inv : 0.0f;
}

// ---------------- host orchestration ----------------
extern "C" void kernel_launch(void* const* d_in, const int* in_sizes, int n_in,
                              void* d_out, int out_size, void* d_ws, size_t ws_size,
                              hipStream_t stream) {
  (void)in_sizes; (void)n_in; (void)out_size; (void)ws_size;
  const float* x    = (const float*)d_in[0];
  // d_in[1] = mask (bool) -- causal triu, applied analytically
  const float* cosT = (const float*)d_in[2];
  const float* sinT = (const float*)d_in[3];
  const float* Wq   = (const float*)d_in[4];
  const float* Wk   = (const float*)d_in[5];
  const float* Wv   = (const float*)d_in[6];
  const float* Wout = (const float*)d_in[7];

  float* out = (float*)d_out;                        // [S, OUTD]
  float* p   = out + (size_t)S * OUTD;               // [H, S, S]

  // workspace carve-up (all counts multiples of 1024 -> alignment preserved)
  char* w = (char*)d_ws;
  auto aus = [&](size_t n) { unsigned short* r = (unsigned short*)w; w += n * 2; return r; };
  auto af  = [&](size_t n) { float* r = (float*)w; w += n * 4; return r; };
  unsigned short* xb    = aus((size_t)S * E);       // 4M
  unsigned short* Wqb   = aus((size_t)E * QDIM);    // 4M
  unsigned short* Wkb   = aus((size_t)E * KVD);     // 1M
  unsigned short* Wvb   = aus((size_t)E * KVD);     // 1M
  unsigned short* Woutb = aus((size_t)QDIM * OUTD); // 4M
  unsigned short* Qb    = aus((size_t)S * QDIM);    // 4M
  unsigned short* Kb    = aus((size_t)S * KVD);     // 1M
  unsigned short* Vb    = aus((size_t)S * KVD);     // 1M
  unsigned short* Ob16  = aus((size_t)S * QDIM);    // 4M
  float* Q32 = af((size_t)S * QDIM);                // 4M f32
  float* K32 = af((size_t)S * KVD);                 // 1M f32
  float* V32 = af((size_t)S * KVD);                 // 1M f32
  float* O32 = af((size_t)S * QDIM);                // 4M f32

  dim3 blk(256);
  auto cvt = [&](const float* s, unsigned short* d, size_t n) {
    cvt_f32_to_bf16<<<dim3((unsigned)(n / 1024)), blk, 0, stream>>>(s, d, (int)(n / 4));
  };

  // 1) convert inputs/weights to bf16
  cvt(x,    xb,    (size_t)S * E);
  cvt(Wq,   Wqb,   (size_t)E * QDIM);
  cvt(Wk,   Wkb,   (size_t)E * KVD);
  cvt(Wv,   Wvb,   (size_t)E * KVD);
  cvt(Wout, Woutb, (size_t)QDIM * OUTD);

  // 2) QKV projections (f32 accum)
  gemm_wmma<false, false><<<dim3(QDIM / 128, S / 128, 1), blk, 0, stream>>>(
      xb, Wqb, Q32, E, E, QDIM, QDIM, 0, 0, 0, 0);
  gemm_wmma<false, false><<<dim3(KVD / 128, S / 128, 1), blk, 0, stream>>>(
      xb, Wkb, K32, E, E, KVD, KVD, 0, 0, 0, 0);
  gemm_wmma<false, false><<<dim3(KVD / 128, S / 128, 1), blk, 0, stream>>>(
      xb, Wvb, V32, E, E, KVD, KVD, 0, 0, 0, 0);

  // 3) RoPE(Q), RoPE(K) -> bf16 ; V -> bf16
  rope_bf16<<<dim3((S * QDIM) / 256), blk, 0, stream>>>(Q32, cosT, sinT, Qb, H);
  rope_bf16<<<dim3((S * KVD) / 256), blk, 0, stream>>>(K32, cosT, sinT, Kb, G);
  cvt(V32, Vb, (size_t)S * KVD);

  // 4) scores[h] = Q_h (S x HD) @ K_{h>>2}^T (HD x S) -> p region of d_out
  gemm_wmma<false, true><<<dim3(S / 128, S / 128, H), blk, 0, stream>>>(
      Qb, Kb, p, HD, QDIM, KVD, S,
      /*strideA=*/HD, /*strideB=*/HD, /*strideC=*/(long long)S * S, /*shiftB=*/2);

  // 5) causal masked softmax (scale 1/sqrt(HD)) in place
  softmax_mask_kernel<<<dim3(S, H), blk, 0, stream>>>(p);

  // 6) O[:, h*HD:+HD] = P_h (S x S, f32->bf16 in LDS) @ V_{h>>2} (S x HD)
  gemm_wmma<true, false><<<dim3(HD / 128, S / 128, H), blk, 0, stream>>>(
      p, Vb, O32, S, S, KVD, QDIM,
      /*strideA=*/(long long)S * S, /*strideB=*/HD, /*strideC=*/HD, /*shiftB=*/2);

  // 7) out = O (S x QDIM) @ Wout (QDIM x OUTD)
  cvt(O32, Ob16, (size_t)S * QDIM);
  gemm_wmma<false, false><<<dim3(OUTD / 128, S / 128, 1), blk, 0, stream>>>(
      Ob16, Woutb, out, QDIM, QDIM, OUTD, OUTD, 0, 0, 0, 0);
}
